// ConvLayer_16320875725528
// MI455X (gfx1250) — matrix operand order repaired and checked by reading, not compiled, not run on hardware
//
#include <hip/hip_runtime.h>
#include <hip/hip_bf16.h>

#define NATOMS   100000
#define MNBR     12
#define FNBR     41
#define KDIM     169
#define KP       192          // K padded to 6 steps of 32
#define ROWS_WG  96           // 8 atoms * 12 neighbors
#define ATOMS_WG 8
#define THREADS  192          // 6 waves of 32
#define NROWS    (NATOMS * MNBR)      // 1,200,000
#define NWG      (NROWS / ROWS_WG)    // 12,500 exactly
#define EPSBN    1e-5f

typedef __attribute__((ext_vector_type(16))) __bf16 v16bf;
typedef __attribute__((ext_vector_type(8)))  __bf16 v8bf;
typedef __attribute__((ext_vector_type(8)))  float  v8f;

// ---- workspace layout (float offsets) ----
#define WS_SUM1   0         // [128] BN1 sum
#define WS_SS1    128       // [128] BN1 sumsq
#define WS_SC1    256       // [128] BN1 scale
#define WS_SH1    384       // [128] BN1 shift (linear bias folded in)
#define WS_SUM2   512       // [64]
#define WS_SS2    576       // [64]
#define WS_SC2    640       // [64]
#define WS_SH2    704       // [64]
#define WS_NBR    1024      // [NATOMS*64] f32 pre-BN2 nbr_sumed
// bf16 pre-converted images (optional, used when ws is large enough)
#define WS_WBF    6401024   // [128*192] bf16 (padded W)
#define WS_ABF    6413312   // [NATOMS*64] bf16 atom_fea
#define WS_EBF    9613312   // [NROWS*41] bf16 nbr_fea
#define WS_FULL_FLOATS 34213312ull
#define WS_BASE_FLOATS 6401024ull

__device__ __forceinline__ float softplusf_(float x) {
    return fmaxf(x, 0.f) + log1pf(__expf(-fabsf(x)));
}
__device__ __forceinline__ float sigmoidf_(float x) {
    return 1.f / (1.f + __expf(-x));
}

// A fragment (16-bit A 16x32 layout): lane holds K [kb..kb+7] and [kb+16..kb+23]
__device__ __forceinline__ v16bf frag_A(const __bf16* rowp, int kb) {
    v8bf lo = *reinterpret_cast<const v8bf*>(rowp + kb);
    v8bf hi = *reinterpret_cast<const v8bf*>(rowp + kb + 16);
    return __builtin_shufflevector(lo, hi, 0,1,2,3,4,5,6,7,8,9,10,11,12,13,14,15);
}
// B fragment: lane holds K [kb..kb+7] and [kb+16..kb+23] of its column
__device__ __forceinline__ v16bf frag_B(const __bf16* colp, int kb) {
    v8bf lo = *reinterpret_cast<const v8bf*>(colp + kb);
    v8bf hi = *reinterpret_cast<const v8bf*>(colp + kb + 16);
    return __builtin_shufflevector(lo, hi, 0,1,2,3,4,5,6,7,8,9,10,11,12,13,14,15);
}

// Shared GEMM core. PRE=true: stage from pre-converted bf16 images (vector
// copies); PRE=false: stage from f32 inputs with on-the-fly conversion.
// Wave wv owns rows [16wv,16wv+16) x all 128 cols.
// acc[t] covers output col 64*(t>>2) + 16*(t&3) + (lane&15).
template<bool PRE>
__device__ __forceinline__ void gemm_tile(const float*  __restrict__ atomF,
                                          const float*  __restrict__ nbrF,
                                          const __bf16* __restrict__ atomB,
                                          const __bf16* __restrict__ edgeB,
                                          const __bf16* __restrict__ Wbf,
                                          const float*  __restrict__ Wf,
                                          const int*    __restrict__ nbr_idx,
                                          int rowBase,
                                          __bf16* sA, __bf16* sB,
                                          v8f acc[8])
{
    const int tid  = threadIdx.x;
    const int lane = tid & 31;
    const int wv   = tid >> 5;

    // ---- stage A: [96 rows][192 K] = concat(self, gathered, edge, pad)
    if (PRE) {
        // self + gathered: 16 chunks of 8 bf16 per row, pure b128 copies
        for (int e = tid; e < ROWS_WG * 16; e += THREADS) {
            int r = e >> 4, c = e & 15;
            int gr = rowBase + r;
            int n = gr / MNBR, m = gr - n * MNBR;
            const __bf16* src;
            if (c < 8) {
                src = atomB + n * 64 + c * 8;
            } else {
                int nb = nbr_idx[n * MNBR + m];
                src = atomB + nb * 64 + (c - 8) * 8;
            }
            *reinterpret_cast<v8bf*>(&sA[r * KP + c * 8]) =
                *reinterpret_cast<const v8bf*>(src);
        }
        // edge (41 feats) + zero pad: per-element (rows of 41 are unaligned)
        for (int e = tid; e < ROWS_WG * 64; e += THREADS) {
            int r = e >> 6, f = e & 63;
            int gr = rowBase + r;
            int n = gr / MNBR, m = gr - n * MNBR;
            __bf16 v = (__bf16)0.f;
            if (f < FNBR) v = edgeB[(n * MNBR + m) * FNBR + f];
            sA[r * KP + 128 + f] = v;
        }
    } else {
        for (int e = tid; e < ROWS_WG * KP; e += THREADS) {
            int r = e / KP, f = e - r * KP;
            int gr = rowBase + r;
            int n = gr / MNBR, m = gr - n * MNBR;
            float v = 0.f;
            if (f < 64) {
                v = atomF[n * 64 + f];
            } else if (f < 128) {
                int nb = nbr_idx[n * MNBR + m];
                v = atomF[nb * 64 + (f - 64)];
            } else if (f < KDIM) {
                v = nbrF[(n * MNBR + m) * FNBR + (f - 128)];
            }
            sA[e] = (__bf16)v;
        }
    }

    const int mrow = lane & 15;
    const int kbA  = (lane < 16) ? 0 : 8;
    const int kbB  = (lane < 16) ? 0 : 8;
    const __bf16* aRow = sA + (wv * 16 + mrow) * KP;

    #pragma unroll
    for (int t = 0; t < 8; ++t)
        #pragma unroll
        for (int j = 0; j < 8; ++j) acc[t][j] = 0.f;

    // B staged in two 64-column halves to stay under 64 KB LDS
    for (int h = 0; h < 2; ++h) {
        __syncthreads();
        if (PRE) {
            for (int e = tid; e < 64 * (KP / 8); e += THREADS) {
                int c = e / 24, ch = e - c * 24;
                *reinterpret_cast<v8bf*>(&sB[c * KP + ch * 8]) =
                    *reinterpret_cast<const v8bf*>(&Wbf[(c + 64 * h) * KP + ch * 8]);
            }
        } else {
            for (int e = tid; e < 64 * KP; e += THREADS) {
                int c = e / KP, f = e - c * KP;
                float v = (f < KDIM) ? Wf[(c + 64 * h) * KDIM + f] : 0.f;
                sB[e] = (__bf16)v;
            }
        }
        __syncthreads();
        if (PRE && h == 0 && tid < 64) {
            // pull second W half toward the caches while half 0 computes
            __builtin_prefetch(&Wbf[(64 + tid) * KP], 0, 3);   // global_prefetch_b8
        }
        for (int ks = 0; ks < 6; ++ks) {
            v16bf af = frag_A(aRow, ks * 32 + kbA);
            #pragma unroll
            for (int t = 0; t < 4; ++t) {
                v16bf bfv = frag_B(sB + (t * 16 + mrow) * KP, ks * 32 + kbB);
                acc[h * 4 + t] = __builtin_amdgcn_wmma_f32_16x16x32_bf16(
                    false, af, false, bfv, (short)0, acc[h * 4 + t], false, false);
            }
        }
    }
}

// ---- prep kernels: one-time f32 -> bf16 conversion into ws ----
__global__ void kprep_W(const float* __restrict__ Wf, __bf16* __restrict__ Wbf)
{
    for (int e = blockIdx.x * 256 + threadIdx.x; e < 128 * KP; e += gridDim.x * 256) {
        int c = e / KP, f = e - c * KP;
        Wbf[e] = (f < KDIM) ? (__bf16)Wf[c * KDIM + f] : (__bf16)0.f;
    }
}
__global__ void kprep_cvt(const float* __restrict__ src, __bf16* __restrict__ dst, int n)
{
    int i = blockIdx.x * 256 + threadIdx.x;
    if (i < n) dst[i] = (__bf16)src[i];
}

// ---- Pass 1: GEMM + per-channel BN1 sum/sumsq (bias added before stats) ----
template<bool PRE>
__global__ __launch_bounds__(THREADS)
void k1_gemm_stats(const float* __restrict__ atomF,
                   const float* __restrict__ nbrF,
                   const int*   __restrict__ nbr_idx,
                   const float* __restrict__ Wf,
                   const float* __restrict__ bias,
                   float* __restrict__ ws)
{
    __shared__ __bf16 sA[ROWS_WG * KP];
    __shared__ __bf16 sB[64 * KP];
    __shared__ float  sStat[256];
    for (int e = threadIdx.x; e < 256; e += THREADS) sStat[e] = 0.f;

    const __bf16* atomB = reinterpret_cast<const __bf16*>(ws + WS_ABF);
    const __bf16* edgeB = reinterpret_cast<const __bf16*>(ws + WS_EBF);
    const __bf16* Wbf   = reinterpret_cast<const __bf16*>(ws + WS_WBF);

    v8f acc[8];
    gemm_tile<PRE>(atomF, nbrF, atomB, edgeB, Wbf, Wf, nbr_idx,
                   blockIdx.x * ROWS_WG, sA, sB, acc);

    const int lane = threadIdx.x & 31;
    const int mrow = lane & 15;
    #pragma unroll
    for (int t = 0; t < 8; ++t) {
        int col = 64 * (t >> 2) + 16 * (t & 3) + mrow;
        float bv = bias[col];
        float s = 0.f, ss = 0.f;
        #pragma unroll
        for (int j = 0; j < 8; ++j) {
            float v = acc[t][j] + bv;
            s += v; ss += v * v;
        }
        atomicAdd(&sStat[col], s);          // ds_add_f32
        atomicAdd(&sStat[128 + col], ss);
    }
    __syncthreads();
    for (int c = threadIdx.x; c < 128; c += THREADS) {
        unsafeAtomicAdd(&ws[WS_SUM1 + c], sStat[c]);        // global_atomic_add_f32
        unsafeAtomicAdd(&ws[WS_SS1  + c], sStat[128 + c]);
    }
}

__global__ void k_fin1(const float* __restrict__ bias,
                       const float* __restrict__ g1,
                       const float* __restrict__ b1,
                       float* __restrict__ ws)
{
    int c = threadIdx.x;                          // 128 threads
    float inv  = 1.f / (float)NROWS;
    float mean = ws[WS_SUM1 + c] * inv;
    float var  = ws[WS_SS1 + c] * inv - mean * mean;
    float sc   = g1[c] * rsqrtf(var + EPSBN);
    ws[WS_SC1 + c] = sc;
    ws[WS_SH1 + c] = (bias[c] - mean) * sc + b1[c];   // fold linear bias into BN shift
}

// ---- Pass 2: recompute GEMM, BN1 affine, sigmoid(filter)*softplus(core),
// sum over M per atom, write pre-BN2 nbr_sumed + BN2 stats. ----
template<bool PRE>
__global__ __launch_bounds__(THREADS)
void k2_gemm_act(const float* __restrict__ atomF,
                 const float* __restrict__ nbrF,
                 const int*   __restrict__ nbr_idx,
                 const float* __restrict__ Wf,
                 float* __restrict__ ws)
{
    __shared__ __bf16 sA[ROWS_WG * KP];
    __shared__ __bf16 sB[64 * KP];
    __shared__ float  sSum[ATOMS_WG * 64];
    for (int e = threadIdx.x; e < ATOMS_WG * 64; e += THREADS) sSum[e] = 0.f;

    const __bf16* atomB = reinterpret_cast<const __bf16*>(ws + WS_ABF);
    const __bf16* edgeB = reinterpret_cast<const __bf16*>(ws + WS_EBF);
    const __bf16* Wbf   = reinterpret_cast<const __bf16*>(ws + WS_WBF);

    v8f acc[8];
    const int rowBase = blockIdx.x * ROWS_WG;
    gemm_tile<PRE>(atomF, nbrF, atomB, edgeB, Wbf, Wf, nbr_idx,
                   rowBase, sA, sB, acc);

    const int lane = threadIdx.x & 31;
    const int wv   = threadIdx.x >> 5;
    const int mrow = lane & 15;
    const int rOff = (lane < 16) ? 0 : 8;
    const float* sc1 = ws + WS_SC1;
    const float* sh1 = ws + WS_SH1;

    #pragma unroll
    for (int t = 0; t < 4; ++t) {
        int cf = 16 * t + mrow;            // filter channel (acc[t])
        int cc = 64 + cf;                  // paired core channel (acc[4+t])
        float aF = sc1[cf], bF = sh1[cf];
        float aC = sc1[cc], bC = sh1[cc];
        #pragma unroll
        for (int j = 0; j < 8; ++j) {
            float vf = acc[t][j] * aF + bF;
            float vc = acc[4 + t][j] * aC + bC;
            float prod = sigmoidf_(vf) * softplusf_(vc);
            int rloc = wv * 16 + rOff + j;     // row within workgroup
            int aLoc = rloc / MNBR;            // local atom 0..7
            atomicAdd(&sSum[aLoc * 64 + cf], prod);   // sum over M via ds_add_f32
        }
    }
    __syncthreads();
    if (threadIdx.x < 64) {
        int c = threadIdx.x;
        float s = 0.f, ss = 0.f;
        #pragma unroll
        for (int a = 0; a < ATOMS_WG; ++a) {
            float v = sSum[a * 64 + c];
            s += v; ss += v * v;
            ws[WS_NBR + (size_t)(rowBase / MNBR + a) * 64 + c] = v;
        }
        unsafeAtomicAdd(&ws[WS_SUM2 + c], s);
        unsafeAtomicAdd(&ws[WS_SS2  + c], ss);
    }
}

__global__ void k_fin2(const float* __restrict__ g2,
                       const float* __restrict__ b2,
                       float* __restrict__ ws)
{
    int c = threadIdx.x;                          // 64 threads
    float inv  = 1.f / (float)NATOMS;
    float mean = ws[WS_SUM2 + c] * inv;
    float var  = ws[WS_SS2 + c] * inv - mean * mean;
    float sc   = g2[c] * rsqrtf(var + EPSBN);
    ws[WS_SC2 + c] = sc;
    ws[WS_SH2 + c] = b2[c] - mean * sc;
}

__global__ void k3_out(const float* __restrict__ atomF,
                       const float* __restrict__ ws,
                       float* __restrict__ out)
{
    int i = blockIdx.x * 256 + threadIdx.x;
    if (i < NATOMS * 64) {
        int c = i & 63;
        float x = atomF[i] + ws[WS_NBR + i] * ws[WS_SC2 + c] + ws[WS_SH2 + c];
        out[i] = softplusf_(x);
    }
}

__global__ void k0_zero(float* __restrict__ ws)
{
    for (int i = threadIdx.x; i < 768; i += 256) ws[i] = 0.f;
}

extern "C" void kernel_launch(void* const* d_in, const int* in_sizes, int n_in,
                              void* d_out, int out_size, void* d_ws, size_t ws_size,
                              hipStream_t stream)
{
    const float* atom = (const float*)d_in[0];
    const float* nbr  = (const float*)d_in[1];
    const int*   idx  = (const int*)  d_in[2];   // integer input -> int32 per harness
    const float* Wm   = (const float*)d_in[3];
    const float* b    = (const float*)d_in[4];
    const float* g1   = (const float*)d_in[5];
    const float* b1   = (const float*)d_in[6];
    const float* g2   = (const float*)d_in[7];
    const float* b2   = (const float*)d_in[8];
    float* out = (float*)d_out;
    float* ws  = (float*)d_ws;

    const bool pre = ws_size >= WS_FULL_FLOATS * sizeof(float);

    k0_zero<<<1, 256, 0, stream>>>(ws);
    if (pre) {
        __bf16* Wbf   = reinterpret_cast<__bf16*>(ws + WS_WBF);
        __bf16* atomB = reinterpret_cast<__bf16*>(ws + WS_ABF);
        __bf16* edgeB = reinterpret_cast<__bf16*>(ws + WS_EBF);
        kprep_W<<<(128 * KP + 255) / 256, 256, 0, stream>>>(Wm, Wbf);
        kprep_cvt<<<(NATOMS * 64 + 255) / 256, 256, 0, stream>>>(atom, atomB, NATOMS * 64);
        kprep_cvt<<<(NROWS * FNBR + 255) / 256, 256, 0, stream>>>(nbr, edgeB, NROWS * FNBR);
        k1_gemm_stats<true><<<NWG, THREADS, 0, stream>>>(atom, nbr, idx, Wm, b, ws);
        k_fin1<<<1, 128, 0, stream>>>(b, g1, b1, ws);
        k2_gemm_act<true><<<NWG, THREADS, 0, stream>>>(atom, nbr, idx, Wm, ws);
    } else {
        k1_gemm_stats<false><<<NWG, THREADS, 0, stream>>>(atom, nbr, idx, Wm, b, ws);
        k_fin1<<<1, 128, 0, stream>>>(b, g1, b1, ws);
        k2_gemm_act<false><<<NWG, THREADS, 0, stream>>>(atom, nbr, idx, Wm, ws);
    }
    k_fin2<<<1, 64, 0, stream>>>(g2, b2, ws);
    k3_out<<<(NATOMS * 64 + 255) / 256, 256, 0, stream>>>(atom, ws, out);
}
